// WindowAttention_71786083385543
// MI455X (gfx1250) — matrix operand order repaired and checked by reading, not compiled, not run on hardware
//
#include <hip/hip_runtime.h>
#include <hip/hip_bf16.h>

// ---------------------------------------------------------------------------
// Fused Swin window attention for gfx1250 (MI455X), wave32 + WMMA bf16.
//   B = 4096 windows, N = 64 tokens, DIM = 384, HEADS = 12, HD = 32.
// One block per window, 384 threads = 12 waves (1 wave / head in attention).
// All GEMMs use v_wmma_f32_16x16x32_bf16 (fp32 accumulate).
// ---------------------------------------------------------------------------

typedef __attribute__((ext_vector_type(16))) __bf16 v16bf;
typedef __attribute__((ext_vector_type(8)))  float  v8f;
typedef __attribute__((ext_vector_type(2)))  __bf16 bf16x2;

#define NTOK   64
#define DIM    384
#define HEADS  12
#define HD     32
#define QKVN   1152
#define SCALE  0.17677669529663687f   // 1/sqrt(32)

// packed weight tile counts (tile = 32K x 16N bf16 = 512 elements)
#define QKV_KT  12     // 384/32
#define QKV_NT  72     // 1152/16
#define PRJ_KT  12
#define PRJ_NT  24     // 384/16
#define QKV_ELEMS (QKV_KT * QKV_NT * 512)   // 442368
#define PRJ_ELEMS (PRJ_KT * PRJ_NT * 512)   // 147456

static __device__ __forceinline__ v8f wmma_bf16(v16bf a, v16bf b, v8f c) {
  return __builtin_amdgcn_wmma_f32_16x16x32_bf16(false, a, false, b, (short)0, c,
                                                 false, false);
}

static __device__ __forceinline__ unsigned pack_bf2(float a, float b) {
  bf16x2 p; p[0] = (__bf16)a; p[1] = (__bf16)b;
  return __builtin_bit_cast(unsigned, p);
}

// A-operand 16x32 bf16 tile from row-major [rows][ld] LDS matrix.
// Order used here (contiguous k): lanes 0-15 hold M=idx, K = 0..15; lanes 16-31: K = 16..31.
static __device__ __forceinline__ v16bf load_tileA(const __bf16* src, int ld,
                                                   int row0, int col0, int lane) {
  int hl = lane >> 4, idx = lane & 15;
  const __bf16* r = src + (row0 + idx) * ld + col0 + 16 * hl;
  v16bf a;
#pragma unroll
  for (int v = 0; v < 8; ++v) {
    int c = 2 * v;                              // 0,2,4,...,14
    unsigned u = *(const unsigned*)(r + c);     // contiguous K pair
    bf16x2 p = __builtin_bit_cast(bf16x2, u);
    a[2 * v] = p[0]; a[2 * v + 1] = p[1];
  }
  return a;
}

// B-operand 32x16 tile, source stored as B^T row-major: src[n][k], ld = K stride.
// Order used here (contiguous k, as load_tileA): lane n holds column n; lanes 0-15: K=0..15, lanes 16-31: K=16..31.
static __device__ __forceinline__ v16bf load_tileB(const __bf16* src, int ld,
                                                   int n0, int k0, int lane) {
  int hl = lane >> 4, idx = lane & 15;
  const __bf16* r = src + (n0 + idx) * ld + k0 + 16 * hl;
  v16bf b;
#pragma unroll
  for (int v = 0; v < 8; ++v) {
    unsigned u = *(const unsigned*)(r + 2 * v);
    bf16x2 p = __builtin_bit_cast(bf16x2, u);
    b[2 * v] = p[0]; b[2 * v + 1] = p[1];
  }
  return b;
}

// B-operand tile from pre-packed weights: lane's 16 values contiguous (32 B).
static __device__ __forceinline__ v16bf load_tileB_packed(const __bf16* tile,
                                                          int lane) {
  const unsigned* u = (const unsigned*)(tile + lane * 16);
  v16bf b;
#pragma unroll
  for (int v = 0; v < 8; ++v) {
    bf16x2 p = __builtin_bit_cast(bf16x2, u[v]);
    b[2 * v] = p[0]; b[2 * v + 1] = p[1];
  }
  return b;
}

// ---------------------------------------------------------------------------
// Weight repack: fp32 -> bf16 tiles in B-operand register order.
// Tile order [nt][kt]; within tile: lane-major (32 lanes x 16 elems),
// element j of lane L maps to k_local = j + 16*(L>>4), n_local = L&15.
// ---------------------------------------------------------------------------
__global__ void repack_weights(const float* __restrict__ qkv_w,
                               const float* __restrict__ proj_w,
                               __bf16* __restrict__ ws) {
  int e = blockIdx.x * blockDim.x + threadIdx.x;
  if (e < QKV_ELEMS) {
    int tile = e >> 9, lane = (e >> 4) & 31, j = e & 15;
    int nt = tile / QKV_KT, kt = tile % QKV_KT;
    int n = nt * 16 + (lane & 15);
    int k = kt * 32 + j + 16 * (lane >> 4);
    ws[e] = (__bf16)qkv_w[k * QKVN + n];
  } else if (e < QKV_ELEMS + PRJ_ELEMS) {
    int e2 = e - QKV_ELEMS;
    int tile = e2 >> 9, lane = (e2 >> 4) & 31, j = e2 & 15;
    int nt = tile / PRJ_KT, kt = tile % PRJ_KT;
    int n = nt * 16 + (lane & 15);
    int k = kt * 32 + j + 16 * (lane >> 4);
    ws[QKV_ELEMS + e2] = (__bf16)proj_w[k * DIM + n];
  }
}

// ---------------------------------------------------------------------------
// Fused attention kernel. Dynamic LDS = 288 KB:
//   sQ  [12][64][32] bf16   48 KB   (q, pre-scaled)
//   sK  [12][64][32] bf16   48 KB
//   sVT [12][32][64] bf16   48 KB   (v transposed)
//   sP  [12][64][64] bf16   96 KB   (scores -> probs; overlays sX in phase A)
//   sO  [64][384]    bf16   48 KB   (concat-head attention output)
// ---------------------------------------------------------------------------
__global__ __launch_bounds__(384, 1)
void win_attn_kernel(const float* __restrict__ x, const float* __restrict__ mask,
                     const float* __restrict__ qkv_b, const float* __restrict__ rpb,
                     const float* __restrict__ proj_b, const int* __restrict__ rpi,
                     const __bf16* __restrict__ wpk, float* __restrict__ out) {
  extern __shared__ char smem[];
  __bf16* sQ  = (__bf16*)smem;                       // 12*64*32 = 24576 elems
  __bf16* sK  = sQ  + HEADS * NTOK * HD;
  __bf16* sVT = sK  + HEADS * NTOK * HD;
  __bf16* sP  = sVT + HEADS * NTOK * HD;             // 12*64*64 = 49152 elems
  __bf16* sX  = sP;                                  // overlay, 24576 elems
  __bf16* sO  = sP  + HEADS * NTOK * NTOK;           // 64*384   = 24576 elems

  const int b    = blockIdx.x;
  const int tid  = threadIdx.x;
  const int lane = tid & 31;
  // wave index is uniform across the wave: pin it to an SGPR so all tile
  // indices / head ids / q-k-v selectors become scalar (uniform branches).
  const int wave = __builtin_amdgcn_readfirstlane(tid >> 5);   // 0..11
  const int idx  = lane & 15;
  const int hl   = lane >> 4;

  // ---------------- Phase A1: stream x -> bf16 LDS -------------------------
  const float* xg = x + (size_t)b * NTOK * DIM;
  unsigned* sXu = (unsigned*)sX;
  for (int i = tid; i < NTOK * DIM / 4; i += 384) {
    float4 f = ((const float4*)xg)[i];
    sXu[2 * i + 0] = pack_bf2(f.x, f.y);
    sXu[2 * i + 1] = pack_bf2(f.z, f.w);
  }
  __syncthreads();

  // ---------------- Phase A2: qkv = x @ qkv_w + b --------------------------
  // 4 M-tiles x 72 N-tiles = 288 tiles, 24 per wave, 12 k-steps each.
#pragma unroll 1
  for (int i = 0; i < 24; ++i) {
    int tileId = wave * 24 + i;          // scalar
    int mt = tileId & 3, nt = tileId >> 2;
    v8f acc = {};
#pragma unroll 2
    for (int kt = 0; kt < QKV_KT; ++kt) {
      v16bf a  = load_tileA(sX, DIM, mt * 16, kt * 32, lane);
      v16bf bm = load_tileB_packed(wpk + (size_t)(nt * QKV_KT + kt) * 512, lane);
      acc = wmma_bf16(a, bm, acc);
    }
    // scalar decode of output column block: s (q/k/v), head, dim base
    int s    = (nt * 16) / DIM;          // 0,1,2 — uniform
    int rem0 = nt * 16 - s * DIM;        // uniform
    int h2   = rem0 >> 5;                // uniform head
    int d0   = rem0 & 31;                // uniform: 0 or 16
    float bias = qkv_b[nt * 16 + idx];
    int d = d0 + idx;                    // per-lane dim 0..31
    if (s == 0) {
      __bf16* dst = sQ + h2 * (NTOK * HD) + (mt * 16 + 8 * hl) * HD + d;
#pragma unroll
      for (int r = 0; r < 8; ++r) dst[r * HD] = (__bf16)((acc[r] + bias) * SCALE);
    } else if (s == 1) {
      __bf16* dst = sK + h2 * (NTOK * HD) + (mt * 16 + 8 * hl) * HD + d;
#pragma unroll
      for (int r = 0; r < 8; ++r) dst[r * HD] = (__bf16)(acc[r] + bias);
    } else {
      // transposed store: consecutive r -> consecutive tokens (contiguous)
      unsigned* dst = (unsigned*)(sVT + h2 * (NTOK * HD) + d * NTOK +
                                  mt * 16 + 8 * hl);
#pragma unroll
      for (int v = 0; v < 4; ++v)
        dst[v] = pack_bf2(acc[2 * v] + bias, acc[2 * v + 1] + bias);
    }
  }
  __syncthreads();

  // ---------------- Phase B: scores + bias + mask + softmax (wave = head) --
  const int h = wave;
  __bf16* qh = sQ  + h * (NTOK * HD);
  __bf16* kh = sK  + h * (NTOK * HD);
  __bf16* vh = sVT + h * (NTOK * HD);
  __bf16* Ph = sP  + h * (NTOK * NTOK);
  const float* maskb = mask + (size_t)b * NTOK * NTOK;

#pragma unroll 1
  for (int mt = 0; mt < 4; ++mt) {
    v16bf a = load_tileA(qh, HD, mt * 16, 0, lane);   // 16x32, full head dim
#pragma unroll 1
    for (int nt = 0; nt < 4; ++nt) {
      v16bf bm = load_tileB(kh, HD, nt * 16, 0, lane); // k^T tile
      v8f acc = {};
      acc = wmma_bf16(a, bm, acc);
      int n = nt * 16 + idx;
#pragma unroll
      for (int r = 0; r < 8; ++r) {
        int m = mt * 16 + r + 8 * hl;
        float bv = rpb[rpi[m * NTOK + n] * HEADS + h];
        Ph[m * NTOK + n] = (__bf16)(acc[r] + bv + maskb[m * NTOK + n]);
      }
    }
  }
  // row softmax: 2 rows per lane (rows lane, lane+32), wave-local.
#pragma unroll 1
  for (int rr = 0; rr < 2; ++rr) {
    int m = lane + 32 * rr;
    unsigned* row = (unsigned*)(Ph + m * NTOK);        // 32 packed bf16x2
    float mx = -3.0e38f;
#pragma unroll 1
    for (int j = 0; j < NTOK / 2; ++j) {
      bf16x2 p = __builtin_bit_cast(bf16x2, row[j]);
      mx = fmaxf(mx, fmaxf((float)p[0], (float)p[1]));
    }
    float sum = 0.f;
#pragma unroll 1
    for (int j = 0; j < NTOK / 2; ++j) {
      bf16x2 p = __builtin_bit_cast(bf16x2, row[j]);
      float e0 = __expf((float)p[0] - mx);
      float e1 = __expf((float)p[1] - mx);
      sum += e0 + e1;
      row[j] = pack_bf2(e0, e1);
    }
    float inv = 1.f / sum;
#pragma unroll 1
    for (int j = 0; j < NTOK / 2; ++j) {
      bf16x2 p = __builtin_bit_cast(bf16x2, row[j]);
      row[j] = pack_bf2((float)p[0] * inv, (float)p[1] * inv);
    }
  }

  // ---------------- Phase C: O_h = P @ v  (via vT) -------------------------
#pragma unroll 1
  for (int mt = 0; mt < 4; ++mt) {
#pragma unroll 1
    for (int nt = 0; nt < 2; ++nt) {
      v8f acc = {};
#pragma unroll
      for (int kt = 0; kt < 2; ++kt) {
        v16bf a  = load_tileA(Ph, NTOK, mt * 16, kt * 32, lane);
        v16bf bm = load_tileB(vh, NTOK, nt * 16, kt * 32, lane);
        acc = wmma_bf16(a, bm, acc);
      }
      __bf16* dst = sO + (mt * 16 + 8 * hl) * DIM + h * HD + nt * 16 + idx;
#pragma unroll
      for (int r = 0; r < 8; ++r) dst[r * DIM] = (__bf16)acc[r];
    }
  }
  __syncthreads();

  // ---------------- Phase D: out = O @ proj_w + proj_b ---------------------
  const __bf16* wp = wpk + QKV_ELEMS;
  float* og = out + (size_t)b * NTOK * DIM;
#pragma unroll 1
  for (int i = 0; i < 8; ++i) {
    int tileId = wave * 8 + i;           // scalar
    int mt = tileId & 3, nt = tileId >> 2;   // nt 0..23
    v8f acc = {};
#pragma unroll 2
    for (int kt = 0; kt < PRJ_KT; ++kt) {
      v16bf a  = load_tileA(sO, DIM, mt * 16, kt * 32, lane);
      v16bf bm = load_tileB_packed(wp + (size_t)(nt * PRJ_KT + kt) * 512, lane);
      acc = wmma_bf16(a, bm, acc);
    }
    float pb = proj_b[nt * 16 + idx];
    float* dst = og + (mt * 16 + 8 * hl) * DIM + nt * 16 + idx;
#pragma unroll
    for (int r = 0; r < 8; ++r) dst[r * DIM] = acc[r] + pb;
  }
}

// ---------------------------------------------------------------------------
extern "C" void kernel_launch(void* const* d_in, const int* in_sizes, int n_in,
                              void* d_out, int out_size, void* d_ws, size_t ws_size,
                              hipStream_t stream) {
  const float* x      = (const float*)d_in[0];
  const float* mask   = (const float*)d_in[1];
  const float* qkv_w  = (const float*)d_in[2];
  const float* qkv_b  = (const float*)d_in[3];
  const float* rpb    = (const float*)d_in[4];
  const float* proj_w = (const float*)d_in[5];
  const float* proj_b = (const float*)d_in[6];
  const int*   rpi    = (const int*)d_in[7];
  float* out = (float*)d_out;

  __bf16* wpk = (__bf16*)d_ws;

  // one-time-per-call weight repack (L2-resident afterwards)
  int repack_elems = QKV_ELEMS + PRJ_ELEMS;
  repack_weights<<<(repack_elems + 255) / 256, 256, 0, stream>>>(qkv_w, proj_w, wpk);

  const int smem_bytes = 288 * 1024;   // 288 KB dynamic LDS (<= 320 KB/WGP)
  (void)hipFuncSetAttribute((const void*)win_attn_kernel,
                            hipFuncAttributeMaxDynamicSharedMemorySize, smem_bytes);
  win_attn_kernel<<<4096, 384, smem_bytes, stream>>>(
      x, mask, qkv_b, rpb, proj_b, rpi, (const __bf16*)wpk, out);
}
